// CMultiHeadAttention_28595892257163
// MI455X (gfx1250) — compile-verified, matrix operand order checked by reading, not run on hardware
//
#include <hip/hip_runtime.h>
#include <math.h>

#define B_  2
#define L_  2048
#define D_  512
#define H_  8
#define DK_ 64
#define DV_ 64

typedef __attribute__((ext_vector_type(16))) _Float16 v16h;
typedef __attribute__((ext_vector_type(8)))  float    v8f;

struct FragH {
  union { v16h v; _Float16 h[16]; unsigned int w[8]; uint4 q[2]; };
};

#define WMMA_F32_F16(a, b, c) \
  __builtin_amdgcn_wmma_f32_16x16x32_f16(false, (a), false, (b), (short)0, (c), false, false)

// --- fragment loaders (CDNA5 16-bit A/B layout: lane<16 -> K[0..7],K[16..23];
//     lane>=16 -> K[8..15],K[24..31]; row/col = lane&15) -------------------
__device__ __forceinline__ FragH frag_ld_h(const _Float16* p, int ld, int k0, int lane) {
  FragH f;
  int row = lane & 15, half = (lane >> 4) & 1;
  const _Float16* b = p + (size_t)row * ld + k0 + half * 8;
  f.q[0] = *reinterpret_cast<const uint4*>(b);
  f.q[1] = *reinterpret_cast<const uint4*>(b + 16);
  return f;
}

__device__ __forceinline__ FragH frag_ld_f32(const float* p, size_t ld, int k0, int lane) {
  FragH f;
  int row = lane & 15, half = (lane >> 4) & 1;
  const float* b = p + (size_t)row * ld + k0 + half * 8;
#pragma unroll
  for (int j = 0; j < 8; ++j) { f.h[j] = (_Float16)b[j]; f.h[j + 8] = (_Float16)b[j + 16]; }
  return f;
}

__device__ __forceinline__ FragH frag_neg(const FragH& a) {
  FragH f;
#pragma unroll
  for (int j = 0; j < 8; ++j) f.w[j] = a.w[j] ^ 0x80008000u;
  return f;
}

// --- weight transpose + fp32->f16: wT[n][k] = w[k][n], 512x512 -------------
__global__ __launch_bounds__(256) void wtrans_kernel(const float* __restrict__ w,
                                                     _Float16* __restrict__ wT) {
  int idx = blockIdx.x * 256 + threadIdx.x;
  int k = idx >> 9, n = idx & 511;
  wT[(size_t)n * 512 + k] = (_Float16)w[idx];
}

// --- complex GEMM: (Ar + iAi)[M,K] @ (Br + iBi)[K,N], B pre-transposed f16.
// MODE 0: store f16 row-major (Q/K proj). MODE 1: store f16 head-transposed
// vT[(b*H+h)*DV+d][L] (V proj). MODE 2: store fp32 + residual (O proj). ------
template <int MODE>
__global__ __launch_bounds__(256) void cgemm_kernel(
    const float* __restrict__ Ar, const float* __restrict__ Ai,
    const _Float16* __restrict__ BrT, const _Float16* __restrict__ BiT,
    int M, int N, int K, float alpha,
    _Float16* __restrict__ DrH, _Float16* __restrict__ DiH,
    float* __restrict__ DrF, float* __restrict__ DiF,
    const float* __restrict__ ResR, const float* __restrict__ ResI) {
  int lane = threadIdx.x & 31;
  int wave = blockIdx.x * (blockDim.x >> 5) + (threadIdx.x >> 5);
  int ntn = N >> 4;
  int tm = wave / ntn, tn = wave % ntn;
  if (tm >= (M >> 4)) return;
  const float*    ar = Ar  + (size_t)(tm * 16) * K;
  const float*    ai = Ai  + (size_t)(tm * 16) * K;
  const _Float16* br = BrT + (size_t)(tn * 16) * K;
  const _Float16* bi = BiT + (size_t)(tn * 16) * K;
  v8f accr = {}, acci = {};
  for (int k0 = 0; k0 < K; k0 += 32) {
    FragH fAr  = frag_ld_f32(ar, K, k0, lane);
    FragH fAi  = frag_ld_f32(ai, K, k0, lane);
    FragH fBr  = frag_ld_h(br, K, k0, lane);
    FragH fBi  = frag_ld_h(bi, K, k0, lane);
    FragH fBiN = frag_neg(fBi);
    accr = WMMA_F32_F16(fAr.v, fBr.v,  accr);   // xr@wr
    accr = WMMA_F32_F16(fAi.v, fBiN.v, accr);   // - xi@wi
    acci = WMMA_F32_F16(fAi.v, fBr.v,  acci);   // xi@wr
    acci = WMMA_F32_F16(fAr.v, fBi.v,  acci);   // + xr@wi
  }
  int half = (lane >> 4) & 1;
  int col  = tn * 16 + (lane & 15);
#pragma unroll
  for (int r = 0; r < 8; ++r) {
    int m = tm * 16 + half * 8 + r;       // C/D layout: M = r (+8 for hi half)
    float vr = accr[r] * alpha;
    float vi = acci[r] * alpha;
    if (MODE == 0) {
      DrH[(size_t)m * N + col] = (_Float16)vr;
      DiH[(size_t)m * N + col] = (_Float16)vi;
    } else if (MODE == 1) {
      int b = m >> 11, l = m & (L_ - 1);
      int h = col >> 6, d = col & 63;
      size_t idx = ((size_t)(b * H_ + h) * DV_ + d) * L_ + l;
      DrH[idx] = (_Float16)vr;
      DiH[idx] = (_Float16)vi;
    } else {
      size_t idx = (size_t)m * N + col;
      DrF[idx] = vr + ResR[idx];
      DiF[idx] = vi + ResI[idx];
    }
  }
}

// --- scores: a = |q @ conj(k)^T| with mask, written pre-softmax to d_out ----
__global__ __launch_bounds__(256) void scores_kernel(
    const _Float16* __restrict__ qr, const _Float16* __restrict__ qi,
    const _Float16* __restrict__ kr, const _Float16* __restrict__ ki,
    const int* __restrict__ mask, float* __restrict__ attn) {
  int lane = threadIdx.x & 31;
  int wave = blockIdx.x * 8 + (threadIdx.x >> 5);
  const int NT = L_ / 16;                  // 128
  int bh  = wave / (NT * NT);
  int rem = wave % (NT * NT);
  int tq = rem / NT, tk = rem % NT;
  int b = bh >> 3, h = bh & 7;
  const _Float16* qbr = qr + ((size_t)(b * L_) + tq * 16) * 512 + h * 64;
  const _Float16* qbi = qi + ((size_t)(b * L_) + tq * 16) * 512 + h * 64;
  const _Float16* kbr = kr + ((size_t)(b * L_) + tk * 16) * 512 + h * 64;
  const _Float16* kbi = ki + ((size_t)(b * L_) + tk * 16) * 512 + h * 64;
  v8f sr = {}, si = {};
#pragma unroll
  for (int k0 = 0; k0 < DK_; k0 += 32) {
    FragH aR  = frag_ld_h(qbr, 512, k0, lane);
    FragH aI  = frag_ld_h(qbi, 512, k0, lane);
    FragH bR  = frag_ld_h(kbr, 512, k0, lane);
    FragH bI  = frag_ld_h(kbi, 512, k0, lane);
    FragH bIN = frag_neg(bI);
    sr = WMMA_F32_F16(aR.v, bR.v,  sr);   // qr.kr
    sr = WMMA_F32_F16(aI.v, bI.v,  sr);   // + qi.ki
    si = WMMA_F32_F16(aI.v, bR.v,  si);   // qi.kr
    si = WMMA_F32_F16(aR.v, bIN.v, si);   // - qr.ki
  }
  int half = (lane >> 4) & 1;
  int col  = tk * 16 + (lane & 15);
#pragma unroll
  for (int r = 0; r < 8; ++r) {
    int qrow = tq * 16 + half * 8 + r;
    float xr = sr[r], xi = si[r];
    float a = sqrtf(xr * xr + xi * xi);
    int mv = mask[((size_t)b * L_ + qrow) * L_ + col];
    attn[((size_t)bh * L_ + qrow) * L_ + col] = mv ? a : -1e9f;
  }
}

// --- in-place row softmax over 2048 columns --------------------------------
__global__ __launch_bounds__(256) void softmax_kernel(float* __restrict__ attn) {
  __shared__ float sh[256];
  float* row = attn + (size_t)blockIdx.x * L_;
  int t = threadIdx.x;
  float v[8];
  float mx = -3.4e38f;
#pragma unroll
  for (int j = 0; j < 8; ++j) { v[j] = row[t + j * 256]; mx = fmaxf(mx, v[j]); }
  sh[t] = mx; __syncthreads();
  for (int s = 128; s > 0; s >>= 1) { if (t < s) sh[t] = fmaxf(sh[t], sh[t + s]); __syncthreads(); }
  mx = sh[0]; __syncthreads();
  float sum = 0.f;
#pragma unroll
  for (int j = 0; j < 8; ++j) { v[j] = __expf(v[j] - mx); sum += v[j]; }
  sh[t] = sum; __syncthreads();
  for (int s = 128; s > 0; s >>= 1) { if (t < s) sh[t] += sh[t + s]; __syncthreads(); }
  float inv = 1.f / sh[0];
#pragma unroll
  for (int j = 0; j < 8; ++j) row[t + j * 256] = v[j] * inv;
}

// --- context: ctx = attn @ v (attn fp32 converted to f16 fragments) --------
__global__ __launch_bounds__(256) void context_kernel(
    const float* __restrict__ attn,
    const _Float16* __restrict__ vTr, const _Float16* __restrict__ vTi,
    float* __restrict__ ctxr, float* __restrict__ ctxi) {
  int lane = threadIdx.x & 31;
  int wave = blockIdx.x * 8 + (threadIdx.x >> 5);
  int bh  = wave >> 9;                    // 512 tiles per (b,h)
  int rem = wave & 511;
  int tq = rem >> 2, td = rem & 3;
  const float*    ap = attn + ((size_t)bh * L_ + tq * 16) * L_;
  const _Float16* br = vTr + ((size_t)bh * DV_ + td * 16) * L_;
  const _Float16* bi = vTi + ((size_t)bh * DV_ + td * 16) * L_;
  v8f cr = {}, ci = {};
  for (int k0 = 0; k0 < L_; k0 += 32) {
    FragH fa = frag_ld_f32(ap, L_, k0, lane);
    FragH fr = frag_ld_h(br, L_, k0, lane);
    FragH fi = frag_ld_h(bi, L_, k0, lane);
    cr = WMMA_F32_F16(fa.v, fr.v, cr);
    ci = WMMA_F32_F16(fa.v, fi.v, ci);
  }
  int half = (lane >> 4) & 1;
  int b = bh >> 3, h = bh & 7;
  int d = td * 16 + (lane & 15);
#pragma unroll
  for (int r = 0; r < 8; ++r) {
    int qrow = tq * 16 + half * 8 + r;
    size_t idx = ((size_t)(b * L_) + qrow) * 512 + h * 64 + d;
    ctxr[idx] = cr[r];
    ctxi[idx] = ci[r];
  }
}

// --- complex layernorm: stage 1 partial moments ----------------------------
__global__ __launch_bounds__(256) void ln_partial_kernel(
    const float* __restrict__ yr, const float* __restrict__ yi, float* __restrict__ part) {
  int b   = blockIdx.x >> 7;              // 128 blocks per batch
  int blk = blockIdx.x & 127;
  const float* pr = yr + (size_t)b * (L_ * D_);
  const float* pi = yi + (size_t)b * (L_ * D_);
  int base = blk * 8192 + threadIdx.x;
  float s0 = 0, s1 = 0, s2 = 0, s3 = 0, s4 = 0;
#pragma unroll 4
  for (int j = 0; j < 32; ++j) {
    float a = pr[base + j * 256], c = pi[base + j * 256];
    s0 += a; s1 += c; s2 += a * a; s3 += c * c; s4 += a * c;
  }
  __shared__ float sh[5][256];
  int t = threadIdx.x;
  sh[0][t] = s0; sh[1][t] = s1; sh[2][t] = s2; sh[3][t] = s3; sh[4][t] = s4;
  __syncthreads();
  for (int s = 128; s > 0; s >>= 1) {
    if (t < s) { for (int q = 0; q < 5; ++q) sh[q][t] += sh[q][t + s]; }
    __syncthreads();
  }
  if (t == 0) { for (int q = 0; q < 5; ++q) part[(size_t)blockIdx.x * 5 + q] = sh[q][0]; }
}

// --- stage 2: whitening matrix per batch -----------------------------------
__global__ void ln_finalize_kernel(const float* __restrict__ part, float* __restrict__ stats) {
  if (threadIdx.x != 0) return;
  int b = blockIdx.x;
  float s0 = 0, s1 = 0, s2 = 0, s3 = 0, s4 = 0;
  for (int j = 0; j < 128; ++j) {
    const float* p = part + (size_t)(b * 128 + j) * 5;
    s0 += p[0]; s1 += p[1]; s2 += p[2]; s3 += p[3]; s4 += p[4];
  }
  float n = (float)(L_ * D_);
  float mr = s0 / n, mi = s1 / n;
  float Crr = s2 / n - mr * mr + 1e-6f;
  float Cii = s3 / n - mi * mi + 1e-6f;
  float Cri = s4 / n - mr * mi;
  float s = sqrtf(Crr * Cii - Cri * Cri);
  float t = sqrtf(Crr + Cii + 2.f * s);
  float inv = 1.f / (s * t);
  stats[b * 8 + 0] = mr;
  stats[b * 8 + 1] = mi;
  stats[b * 8 + 2] = (Cii + s) * inv;   // Rrr
  stats[b * 8 + 3] = (Crr + s) * inv;   // Rii
  stats[b * 8 + 4] = -Cri * inv;        // Rri
}

// --- stage 3: whiten + complex affine, write final outputs -----------------
__global__ __launch_bounds__(256) void ln_apply_kernel(
    const float* __restrict__ yr, const float* __restrict__ yi, const float* __restrict__ stats,
    const float* __restrict__ lwr, const float* __restrict__ lwi,
    const float* __restrict__ lbr, const float* __restrict__ lbi,
    float* __restrict__ outr, float* __restrict__ outi) {
  size_t idx = (size_t)blockIdx.x * 256 + threadIdx.x;
  int b = (int)(idx >> 20);               // L_*D_ = 2^20
  int d = (int)(idx & 511);
  const float* st = stats + b * 8;
  float r = yr[idx] - st[0], i = yi[idx] - st[1];
  float nr = st[2] * r + st[4] * i;
  float ni = st[3] * i + st[4] * r;
  outr[idx] = nr * lwr[d] - ni * lwi[d] + lbr[d];
  outi[idx] = nr * lwi[d] + ni * lwr[d] + lbi[d];
}

extern "C" void kernel_launch(void* const* d_in, const int* in_sizes, int n_in,
                              void* d_out, int out_size, void* d_ws, size_t ws_size,
                              hipStream_t stream) {
  const float* q_r  = (const float*)d_in[0];
  const float* q_i  = (const float*)d_in[1];
  const float* k_r  = (const float*)d_in[2];
  const float* k_i  = (const float*)d_in[3];
  const float* v_r  = (const float*)d_in[4];
  const float* v_i  = (const float*)d_in[5];
  const float* w_in[8] = { (const float*)d_in[6],  (const float*)d_in[7],    // wq
                           (const float*)d_in[8],  (const float*)d_in[9],    // wk
                           (const float*)d_in[10], (const float*)d_in[11],   // wv
                           (const float*)d_in[12], (const float*)d_in[13] }; // wo
  const float* ln_wr = (const float*)d_in[14];
  const float* ln_wi = (const float*)d_in[15];
  const float* ln_br = (const float*)d_in[16];
  const float* ln_bi = (const float*)d_in[17];
  const int*   mask  = (const int*)d_in[18];

  const size_t MROWS = (size_t)B_ * L_;        // 4096
  const size_t ACT   = MROWS * 512;            // 2,097,152 elems
  const size_t WELEM = 512 * 512;

  char* ws = (char*)d_ws;
  size_t off = 0;
  auto alloc = [&](size_t bytes) -> void* {
    void* p = ws + off;
    off = (off + bytes + 255) & ~(size_t)255;
    return p;
  };
  _Float16* wT[8];
  for (int j = 0; j < 8; ++j) wT[j] = (_Float16*)alloc(WELEM * 2);
  _Float16* qh_r = (_Float16*)alloc(ACT * 2);
  _Float16* qh_i = (_Float16*)alloc(ACT * 2);
  _Float16* kh_r = (_Float16*)alloc(ACT * 2);
  _Float16* kh_i = (_Float16*)alloc(ACT * 2);
  _Float16* vT_r = (_Float16*)alloc(ACT * 2);
  _Float16* vT_i = (_Float16*)alloc(ACT * 2);
  float* ctx_r = (float*)alloc(ACT * 4);
  float* ctx_i = (float*)alloc(ACT * 4);
  float* y_r   = (float*)alloc(ACT * 4);
  float* y_i   = (float*)alloc(ACT * 4);
  float* part  = (float*)alloc((size_t)B_ * 128 * 5 * 4);
  float* stats = (float*)alloc((size_t)B_ * 8 * 4);

  float* out_r = (float*)d_out;
  float* out_i = out_r + ACT;
  float* attn  = out_r + 2 * ACT;              // [B,H,L,L] region of d_out

  // 1) weights -> transposed f16
  for (int j = 0; j < 8; ++j)
    wtrans_kernel<<<(int)(WELEM / 256), 256, 0, stream>>>(w_in[j], wT[j]);

  // 2) QKV complex projections (one wave per 16x16 tile; 8192 tiles each)
  const int gemm_blocks = (int)((MROWS / 16) * (512 / 16) / 8);  // 1024
  cgemm_kernel<0><<<gemm_blocks, 256, 0, stream>>>(q_r, q_i, wT[0], wT[1],
      (int)MROWS, 512, 512, 0.125f, qh_r, qh_i, nullptr, nullptr, nullptr, nullptr);
  cgemm_kernel<0><<<gemm_blocks, 256, 0, stream>>>(k_r, k_i, wT[2], wT[3],
      (int)MROWS, 512, 512, 1.0f, kh_r, kh_i, nullptr, nullptr, nullptr, nullptr);
  cgemm_kernel<1><<<gemm_blocks, 256, 0, stream>>>(v_r, v_i, wT[4], wT[5],
      (int)MROWS, 512, 512, 1.0f, vT_r, vT_i, nullptr, nullptr, nullptr, nullptr);

  // 3) masked |complex scores| -> d_out attn region (pre-softmax)
  const int score_blocks = B_ * H_ * (L_ / 16) * (L_ / 16) / 8;  // 32768
  scores_kernel<<<score_blocks, 256, 0, stream>>>(qh_r, qh_i, kh_r, kh_i, mask, attn);

  // 4) softmax in place
  softmax_kernel<<<B_ * H_ * L_, 256, 0, stream>>>(attn);

  // 5) ctx = attn @ v
  const int ctx_blocks = B_ * H_ * (L_ / 16) * (DV_ / 16) / 8;   // 1024
  context_kernel<<<ctx_blocks, 256, 0, stream>>>(attn, vT_r, vT_i, ctx_r, ctx_i);

  // 6) output projection + residual (fp32 store)
  cgemm_kernel<2><<<gemm_blocks, 256, 0, stream>>>(ctx_r, ctx_i, wT[6], wT[7],
      (int)MROWS, 512, 512, 1.0f, nullptr, nullptr, y_r, y_i, q_r, q_i);

  // 7) complex layernorm
  ln_partial_kernel<<<B_ * 128, 256, 0, stream>>>(y_r, y_i, part);
  ln_finalize_kernel<<<B_, 32, 0, stream>>>(part, stats);
  ln_apply_kernel<<<(int)(ACT / 256), 256, 0, stream>>>(y_r, y_i, stats,
      ln_wr, ln_wi, ln_br, ln_bi, out_r, out_i);
}
